// GNS_50414326120524
// MI455X (gfx1250) — matrix operand order repaired and checked by reading, not compiled, not run on hardware
//
#include <hip/hip_runtime.h>
#include <cstdint>

// ---------------------------------------------------------------------------
// GNS (graph network simulator) for MI455X / gfx1250.
// All dense GEMMs use v_wmma_f32_16x16x32_f16 (wave32 WMMA), f16 in / f32 acc.
// GEMM tiles are staged into LDS with GLOBAL_LOAD_ASYNC_TO_LDS_B128 (ASYNCcnt)
// when the toolchain exposes the builtin; falls back to load+ds_store.
// ---------------------------------------------------------------------------

typedef __attribute__((ext_vector_type(16))) _Float16 v16h;
typedef __attribute__((ext_vector_type(8)))  _Float16 v8h;
typedef __attribute__((ext_vector_type(8)))  float    v8f;

// Builtin signature (from hipcc diagnostic): async-to-LDS b128 takes
// v4i pointers: AS1 (global) source, AS3 (LDS) destination, imm offset, cpol.
typedef int gns_v4i __attribute__((vector_size(16)));
typedef __attribute__((address_space(1))) gns_v4i* gns_gp4;
typedef __attribute__((address_space(3))) gns_v4i* gns_lp4;

#define GNS_N   4096
#define GNS_K   16
#define GNS_E   (GNS_N * GNS_K)     // 65536
#define GNS_H   128
#define GNS_L   8
#define GNS_DT  0.02f
#define GNS_STEPS 2

#if __has_builtin(__builtin_amdgcn_global_load_async_to_lds_b128)
#define GNS_ASYNC_LDS 1
#else
#define GNS_ASYNC_LDS 0
#endif

__device__ __forceinline__ void gns_wait_asynccnt0() {
#if __has_builtin(__builtin_amdgcn_s_wait_asynccnt)
    __builtin_amdgcn_s_wait_asynccnt(0);
#else
    asm volatile("s_wait_asynccnt 0x0" ::: "memory");
#endif
}

// ===========================================================================
// WMMA GEMM:  Y[M,128] = epilogue( X[M,Kdim] @ W[Kdim,128] + bias )
//   mode 0: ReLU        mode 1: LayerNorm(gamma,beta) (+optional residual)
//   mode 2: identity
// X,W are f16 row-major; Kdim multiple of 32; M multiple of 128.
// Block: 256 threads = 8 waves; wave w owns rows [w*16, w*16+16) of the
// 128x128 block tile and all 8 N-tiles (8 x v8f accumulators).
// ===========================================================================
__global__ __launch_bounds__(256)
void gns_wmma_gemm(const _Float16* __restrict__ X,
                   const _Float16* __restrict__ W,
                   const float* __restrict__ bias,
                   const float* __restrict__ gamma,
                   const float* __restrict__ beta,
                   const float* __restrict__ residual,
                   float* __restrict__ out32,
                   _Float16* __restrict__ out16,
                   int M, int Kdim, int mode)
{
    __shared__ _Float16 Xs[128 * 32];   // 8 KB, row stride 32 halves (64B)
    __shared__ _Float16 Ws[32 * 128];   // 8 KB, row stride 128 halves (256B)

    const int tid  = threadIdx.x;
    const int wave = tid >> 5;
    const int lane = tid & 31;
    const int m0   = blockIdx.x * 128;

    v8f acc[8];
    #pragma unroll
    for (int nt = 0; nt < 8; ++nt)
        #pragma unroll
        for (int r = 0; r < 8; ++r) acc[nt][r] = 0.0f;

    const int ln15 = lane & 15;
    const int kg   = lane >> 4;

    for (int k0 = 0; k0 < Kdim; k0 += 32) {
#if GNS_ASYNC_LDS
        // ---- async fill: X tile (128 rows x 32 halves) directly into LDS.
        // INST_OFFSET applies to BOTH the global and LDS address, so the two
        // 16B halves of each 32B chunk share base pointers. ----
        {
            const int r = tid >> 1;            // 0..127
            const int c = (tid & 1) * 16;      // 0 or 16
            const _Float16* srcX = X + (size_t)(m0 + r) * (size_t)Kdim + (size_t)(k0 + c);
            _Float16* dstX = &Xs[r * 32 + c];
            __builtin_amdgcn_global_load_async_to_lds_b128(
                (gns_gp4)srcX, (gns_lp4)dstX, 0, 0);
            __builtin_amdgcn_global_load_async_to_lds_b128(
                (gns_gp4)srcX, (gns_lp4)dstX, 16, 0);
            if (k0 + 32 < Kdim)
                __builtin_prefetch(srcX + 32, 0, 1);
        }
        // ---- async fill: W tile (32 rows x 128 halves) ----
        {
            const int r = tid >> 3;            // 0..31
            const int c = (tid & 7) * 16;      // 0..112
            const _Float16* srcW = W + (size_t)(k0 + r) * 128 + (size_t)c;
            _Float16* dstW = &Ws[r * 128 + c];
            __builtin_amdgcn_global_load_async_to_lds_b128(
                (gns_gp4)srcW, (gns_lp4)dstW, 0, 0);
            __builtin_amdgcn_global_load_async_to_lds_b128(
                (gns_gp4)srcW, (gns_lp4)dstW, 16, 0);
        }
        gns_wait_asynccnt0();       // this wave's async fills have landed
        __syncthreads();            // all waves' fills visible
#else
        // ---- fallback: global->VGPR->ds_store staging ----
        {
            const int r = tid >> 1;
            const int c = (tid & 1) * 16;
            const uint4* src = reinterpret_cast<const uint4*>(
                X + (size_t)(m0 + r) * (size_t)Kdim + (size_t)(k0 + c));
            uint4 a = src[0], b = src[1];
            if (k0 + 32 < Kdim)
                __builtin_prefetch(X + (size_t)(m0 + r) * (size_t)Kdim + (size_t)(k0 + 32 + c), 0, 1);
            uint4* dst = reinterpret_cast<uint4*>(&Xs[r * 32 + c]);
            dst[0] = a; dst[1] = b;
        }
        {
            const int r = tid >> 3;
            const int c = (tid & 7) * 16;
            const uint4* src = reinterpret_cast<const uint4*>(
                W + (size_t)(k0 + r) * 128 + (size_t)c);
            uint4 a = src[0], b = src[1];
            uint4* dst = reinterpret_cast<uint4*>(&Ws[r * 128 + c]);
            dst[0] = a; dst[1] = b;
        }
        __syncthreads();
#endif

        // ---- A fragment (16x32 f16): lane(l<16)=row l K[0..7,16..23],
        //      lane(l>=16)=row l-16 K[8..15,24..31] ----
        const int mrow = wave * 16 + ln15;
        v16h afrag;
        {
            v8h lo = *reinterpret_cast<const v8h*>(&Xs[mrow * 32 + kg * 8]);
            v8h hi = *reinterpret_cast<const v8h*>(&Xs[mrow * 32 + 16 + kg * 8]);
            #pragma unroll
            for (int i = 0; i < 8; ++i) { afrag[i] = lo[i]; afrag[8 + i] = hi[i]; }
        }
        // ---- B fragments: lane holds W col nt*16+(lane&15), K[kg*8..kg*8+7, 16+kg*8..16+kg*8+7] ----
        #pragma unroll
        for (int nt = 0; nt < 8; ++nt) {
            v16h bfrag;
            #pragma unroll
            for (int i = 0; i < 8; ++i) {
                bfrag[i]     = Ws[(kg * 8 + i) * 128 + nt * 16 + ln15];
                bfrag[8 + i] = Ws[(16 + kg * 8 + i) * 128 + nt * 16 + ln15];
            }
            acc[nt] = __builtin_amdgcn_wmma_f32_16x16x32_f16(
                false, afrag, false, bfrag, (short)0, acc[nt], false, false);
        }
        __syncthreads();
    }

    // ---- epilogue: bias, then ReLU / LayerNorm / identity, in registers ----
    #pragma unroll
    for (int nt = 0; nt < 8; ++nt) {
        float bv = bias[nt * 16 + ln15];
        #pragma unroll
        for (int r = 0; r < 8; ++r) acc[nt][r] += bv;
    }

    if (mode == 1) {
        // LayerNorm: row (kg*8+r) is held by the 16 lanes sharing kg;
        // each lane owns cols {nt*16+ln15}. Reduce across the 16-lane group.
        #pragma unroll
        for (int r = 0; r < 8; ++r) {
            float s = 0.0f, s2 = 0.0f;
            #pragma unroll
            for (int nt = 0; nt < 8; ++nt) { float v = acc[nt][r]; s += v; s2 += v * v; }
            #pragma unroll
            for (int msk = 1; msk < 16; msk <<= 1) {
                s  += __shfl_xor(s,  msk, 32);
                s2 += __shfl_xor(s2, msk, 32);
            }
            float mu  = s  * (1.0f / 128.0f);
            float var = s2 * (1.0f / 128.0f) - mu * mu;
            float rs  = rsqrtf(var + 1e-5f);
            size_t grow = (size_t)(m0 + wave * 16 + kg * 8 + r);
            #pragma unroll
            for (int nt = 0; nt < 8; ++nt) {
                int col = nt * 16 + ln15;
                float y = (acc[nt][r] - mu) * rs * gamma[col] + beta[col];
                if (residual) y += residual[grow * 128 + col];
                if (out32) out32[grow * 128 + col] = y;
                if (out16) out16[grow * 128 + col] = (_Float16)y;
            }
        }
    } else {
        #pragma unroll
        for (int r = 0; r < 8; ++r) {
            size_t grow = (size_t)(m0 + wave * 16 + kg * 8 + r);
            #pragma unroll
            for (int nt = 0; nt < 8; ++nt) {
                int col = nt * 16 + ln15;
                float y = acc[nt][r];
                if (mode == 0) y = fmaxf(y, 0.0f);
                if (out32) out32[grow * 128 + col] = y;
                if (out16) out16[grow * 128 + col] = (_Float16)y;
            }
        }
    }
}

// ===========================================================================
// KNN: one block (256 thr) per row i. d^2 row in LDS, 17 argmin extractions
// (ascending distance, lower index on ties; self forced first, matching
// jax.lax.top_k(-d2, K+1) with d2[i][i]==0).
// ===========================================================================
__global__ __launch_bounds__(256)
void gns_knn(const float* __restrict__ pos,
             int* __restrict__ senders, int* __restrict__ receivers)
{
    __shared__ float d2s[GNS_N];
    __shared__ float redv[256];
    __shared__ int   redi[256];

    const int i   = blockIdx.x;
    const int tid = threadIdx.x;
    const float px = pos[i * 3 + 0];
    const float py = pos[i * 3 + 1];
    const float pz = pos[i * 3 + 2];

    for (int j = tid; j < GNS_N; j += 256) {
        float dx = px - pos[j * 3 + 0];
        float dy = py - pos[j * 3 + 1];
        float dz = pz - pos[j * 3 + 2];
        float d = dx * dx + dy * dy + dz * dz;
        d2s[j] = (j == i) ? -1.0f : d;      // force self selected first
    }
    __syncthreads();

    for (int it = 0; it < GNS_K + 1; ++it) {
        float best = 3.0e38f; int bidx = GNS_N;
        for (int j = tid; j < GNS_N; j += 256) {
            float v = d2s[j];
            if (v < best || (v == best && j < bidx)) { best = v; bidx = j; }
        }
        redv[tid] = best; redi[tid] = bidx;
        __syncthreads();
        for (int s = 128; s > 0; s >>= 1) {
            if (tid < s) {
                float ov = redv[tid + s]; int oi = redi[tid + s];
                if (ov < redv[tid] || (ov == redv[tid] && oi < redi[tid])) {
                    redv[tid] = ov; redi[tid] = oi;
                }
            }
            __syncthreads();
        }
        int win = redi[0];
        if (tid == 0) {
            if (it > 0) receivers[i * GNS_K + (it - 1)] = win;
            d2s[win] = 3.0e38f;
        }
        __syncthreads();
    }
    if (tid < GNS_K) senders[i * GNS_K + tid] = i;
}

// edge_attr[e, 0:5] = [dx,dy,dz,dist,seq_sep], padded to 32 cols with zeros (f16)
__global__ __launch_bounds__(256)
void gns_edge_attr(const float* __restrict__ pos, const float* __restrict__ resn,
                   const int* __restrict__ snd, const int* __restrict__ rcv,
                   _Float16* __restrict__ ea)
{
    int e = blockIdx.x * 256 + threadIdx.x;
    if (e >= GNS_E) return;
    int s = snd[e], r = rcv[e];
    float dx = pos[s * 3 + 0] - pos[r * 3 + 0];
    float dy = pos[s * 3 + 1] - pos[r * 3 + 1];
    float dz = pos[s * 3 + 2] - pos[r * 3 + 2];
    float dist = sqrtf(dx * dx + dy * dy + dz * dz);
    float sep = fminf(fabsf(resn[s] - resn[r]) * 0.2f, 1.0f);
    _Float16* row = ea + (size_t)e * 32;
    row[0] = (_Float16)dx; row[1] = (_Float16)dy; row[2] = (_Float16)dz;
    row[3] = (_Float16)dist; row[4] = (_Float16)sep;
    #pragma unroll
    for (int c = 5; c < 32; ++c) row[c] = (_Float16)0.0f;
}

// xcat[e] = concat(xl[rcv[e]], xl[snd[e]], e16[e])  -- one block per edge
__global__ __launch_bounds__(128)
void gns_concat(const _Float16* __restrict__ xl16, const _Float16* __restrict__ e16,
                const int* __restrict__ snd, const int* __restrict__ rcv,
                _Float16* __restrict__ xcat)
{
    int e = blockIdx.x;
    int c = threadIdx.x;
    int s = snd[e], r = rcv[e];
    _Float16* row = xcat + (size_t)e * 384;
    row[c]       = xl16[(size_t)r * 128 + c];
    row[128 + c] = xl16[(size_t)s * 128 + c];
    row[256 + c] = e16[(size_t)e * 128 + c];
}

// agg[n,h] = sum_{k<16} m[n*16 + h/8, 16*(h%8)+k]   (faithful to view(N,128,K).sum(-1))
__global__ __launch_bounds__(256)
void gns_aggregate(const float* __restrict__ m, _Float16* __restrict__ agg16)
{
    int idx = blockIdx.x * 256 + threadIdx.x;   // n*128 + h
    if (idx >= GNS_N * 128) return;
    int n = idx >> 7, h = idx & 127;
    const float* src = m + ((size_t)(n * 16 + (h >> 3))) * 128 + ((h & 7) << 4);
    float s = 0.0f;
    #pragma unroll
    for (int k = 0; k < 16; ++k) s += src[k];
    agg16[idx] = (_Float16)s;
}

// decoder last layer [128->3] + Euler update (one atom per thread)
__global__ __launch_bounds__(256)
void gns_decode_update(const float* __restrict__ h, const float* __restrict__ W2,
                       const float* __restrict__ b2,
                       float* __restrict__ pos, float* __restrict__ vel)
{
    int n = blockIdx.x * 256 + threadIdx.x;
    if (n >= GNS_N) return;
    float a0 = b2[0], a1 = b2[1], a2 = b2[2];
    const float* hr = h + (size_t)n * 128;
    #pragma unroll 4
    for (int k = 0; k < 128; ++k) {
        float v = hr[k];
        a0 += v * W2[k * 3 + 0];
        a1 += v * W2[k * 3 + 1];
        a2 += v * W2[k * 3 + 2];
    }
    const float dt = GNS_DT, hdt2 = 0.5f * GNS_DT * GNS_DT;
    float vx = vel[n * 3 + 0], vy = vel[n * 3 + 1], vz = vel[n * 3 + 2];
    pos[n * 3 + 0] += vx * dt + a0 * hdt2;
    pos[n * 3 + 1] += vy * dt + a1 * hdt2;
    pos[n * 3 + 2] += vz * dt + a2 * hdt2;
    vel[n * 3 + 0] = vx + a0 * dt;
    vel[n * 3 + 1] = vy + a1 * dt;
    vel[n * 3 + 2] = vz + a2 * dt;
}

// pack f32 [Kdim,128] weights -> f16 [Kpad,128] with zero pad rows
__global__ __launch_bounds__(256)
void gns_pack_w(const float* __restrict__ W, _Float16* __restrict__ Wp, int Kdim, int Kpad)
{
    int idx = blockIdx.x * 256 + threadIdx.x;
    if (idx >= Kpad * 128) return;
    int k = idx >> 7;
    Wp[idx] = (k < Kdim) ? (_Float16)W[idx] : (_Float16)0.0f;
}

__global__ __launch_bounds__(256)
void gns_f32_to_f16(const float* __restrict__ src, _Float16* __restrict__ dst, int n)
{
    int idx = blockIdx.x * 256 + threadIdx.x;
    if (idx < n) dst[idx] = (_Float16)src[idx];
}

// ===========================================================================
// Host side
// ===========================================================================
static inline void launch_gemm(const _Float16* X, const _Float16* W, const float* bias,
                               const float* g, const float* b, const float* res,
                               float* o32, _Float16* o16, int M, int Kdim, int mode,
                               hipStream_t s)
{
    hipLaunchKernelGGL(gns_wmma_gemm, dim3(M / 128), dim3(256), 0, s,
                       X, W, bias, g, b, res, o32, o16, M, Kdim, mode);
}

extern "C" void kernel_launch(void* const* d_in, const int* in_sizes, int n_in,
                              void* d_out, int out_size, void* d_ws, size_t ws_size,
                              hipStream_t stream)
{
    (void)in_sizes; (void)n_in; (void)out_size; (void)ws_size;
    // jax pytree flattening: dicts by sorted key.
    // top: coords, dec, edge_enc, node_enc, proc_edge, proc_node, res_numbers, vels, x
    const float* coords  = (const float*)d_in[0];
    const float* dW0 = (const float*)d_in[1];   // [128,128]
    const float* dW1 = (const float*)d_in[2];   // [128,128]
    const float* dW2 = (const float*)d_in[3];   // [128,3]
    const float* db0 = (const float*)d_in[4];
    const float* db1 = (const float*)d_in[5];
    const float* db2 = (const float*)d_in[6];
    const float* eW0 = (const float*)d_in[7];   // [5,128]
    const float* eW1 = (const float*)d_in[8];
    const float* eW2 = (const float*)d_in[9];
    const float* eb0 = (const float*)d_in[10];
    const float* eb1 = (const float*)d_in[11];
    const float* eb2 = (const float*)d_in[12];
    const float* ebeta = (const float*)d_in[13];
    const float* eg    = (const float*)d_in[14];
    const float* nW0 = (const float*)d_in[15];  // [32,128]
    const float* nW1 = (const float*)d_in[16];
    const float* nW2 = (const float*)d_in[17];
    const float* nb0 = (const float*)d_in[18];
    const float* nb1 = (const float*)d_in[19];
    const float* nb2 = (const float*)d_in[20];
    const float* nbeta = (const float*)d_in[21];
    const float* ng    = (const float*)d_in[22];
    const float* peW0 = (const float*)d_in[23]; // [8,384,128]
    const float* peW1 = (const float*)d_in[24]; // [8,128,128]
    const float* peW2 = (const float*)d_in[25];
    const float* peb0 = (const float*)d_in[26]; // [8,128]
    const float* peb1 = (const float*)d_in[27];
    const float* peb2 = (const float*)d_in[28];
    const float* pebeta = (const float*)d_in[29];
    const float* peg    = (const float*)d_in[30];
    const float* pnW0 = (const float*)d_in[31]; // [8,128,128]
    const float* pnW1 = (const float*)d_in[32];
    const float* pnW2 = (const float*)d_in[33];
    const float* pnb0 = (const float*)d_in[34];
    const float* pnb1 = (const float*)d_in[35];
    const float* pnb2 = (const float*)d_in[36];
    const float* pnbeta = (const float*)d_in[37];
    const float* png    = (const float*)d_in[38];
    const float* resn = (const float*)d_in[39]; // [4096,1]
    const float* vels = (const float*)d_in[40];
    const float* xin  = (const float*)d_in[41]; // [4096,32]

    // ---- workspace bump allocator ----
    char* ws = (char*)d_ws;
    size_t off = 0;
    auto alloc = [&](size_t bytes) -> char* {
        char* p = ws + off;
        off = (off + bytes + 255) & ~(size_t)255;
        return p;
    };
    float*    posB   = (float*)alloc((size_t)GNS_N * 3 * 4);
    float*    velB   = (float*)alloc((size_t)GNS_N * 3 * 4);
    int*      snd    = (int*)alloc((size_t)GNS_E * 4);
    int*      rcv    = (int*)alloc((size_t)GNS_E * 4);
    _Float16* ea16   = (_Float16*)alloc((size_t)GNS_E * 32 * 2);
    _Float16* x16    = (_Float16*)alloc((size_t)GNS_N * 32 * 2);
    float*    xl0_32 = (float*)alloc((size_t)GNS_N * 128 * 4);
    _Float16* xl0_16 = (_Float16*)alloc((size_t)GNS_N * 128 * 2);
    float*    xlA32  = (float*)alloc((size_t)GNS_N * 128 * 4);
    _Float16* xlA16  = (_Float16*)alloc((size_t)GNS_N * 128 * 2);
    float*    xlB32  = (float*)alloc((size_t)GNS_N * 128 * 4);
    _Float16* xlB16  = (_Float16*)alloc((size_t)GNS_N * 128 * 2);
    _Float16* agg16  = (_Float16*)alloc((size_t)GNS_N * 128 * 2);
    _Float16* nh16a  = (_Float16*)alloc((size_t)GNS_N * 128 * 2);
    _Float16* nh16b  = (_Float16*)alloc((size_t)GNS_N * 128 * 2);
    float*    nh32   = (float*)alloc((size_t)GNS_N * 128 * 4);
    _Float16* e16    = (_Float16*)alloc((size_t)GNS_E * 128 * 2);
    _Float16* eh16a  = (_Float16*)alloc((size_t)GNS_E * 128 * 2);
    _Float16* eh16b  = (_Float16*)alloc((size_t)GNS_E * 128 * 2);
    float*    m32    = (float*)alloc((size_t)GNS_E * 128 * 4);
    _Float16* xcat16 = (_Float16*)alloc((size_t)GNS_E * 384 * 2);
    // packed f16 weights
    _Float16* nW0p = (_Float16*)alloc(32 * 128 * 2);
    _Float16* nW1p = (_Float16*)alloc(128 * 128 * 2);
    _Float16* nW2p = (_Float16*)alloc(128 * 128 * 2);
    _Float16* eW0p = (_Float16*)alloc(32 * 128 * 2);    // padded 5->32
    _Float16* eW1p = (_Float16*)alloc(128 * 128 * 2);
    _Float16* eW2p = (_Float16*)alloc(128 * 128 * 2);
    _Float16* peW0p = (_Float16*)alloc((size_t)GNS_L * 384 * 128 * 2);
    _Float16* peW1p = (_Float16*)alloc((size_t)GNS_L * 128 * 128 * 2);
    _Float16* peW2p = (_Float16*)alloc((size_t)GNS_L * 128 * 128 * 2);
    _Float16* pnW0p = (_Float16*)alloc((size_t)GNS_L * 128 * 128 * 2);
    _Float16* pnW1p = (_Float16*)alloc((size_t)GNS_L * 128 * 128 * 2);
    _Float16* pnW2p = (_Float16*)alloc((size_t)GNS_L * 128 * 128 * 2);
    _Float16* dW0p = (_Float16*)alloc(128 * 128 * 2);
    _Float16* dW1p = (_Float16*)alloc(128 * 128 * 2);

    auto pack = [&](const float* Wsrc, _Float16* Wdst, int Kdim, int Kpad) {
        int ntot = Kpad * 128;
        hipLaunchKernelGGL(gns_pack_w, dim3((ntot + 255) / 256), dim3(256), 0, stream,
                           Wsrc, Wdst, Kdim, Kpad);
    };

    // ---- weight packing ----
    pack(nW0, nW0p, 32, 32);  pack(nW1, nW1p, 128, 128); pack(nW2, nW2p, 128, 128);
    pack(eW0, eW0p, 5, 32);   pack(eW1, eW1p, 128, 128); pack(eW2, eW2p, 128, 128);
    for (int l = 0; l < GNS_L; ++l) {
        pack(peW0 + (size_t)l * 384 * 128, peW0p + (size_t)l * 384 * 128, 384, 384);
        pack(peW1 + (size_t)l * 128 * 128, peW1p + (size_t)l * 128 * 128, 128, 128);
        pack(peW2 + (size_t)l * 128 * 128, peW2p + (size_t)l * 128 * 128, 128, 128);
        pack(pnW0 + (size_t)l * 128 * 128, pnW0p + (size_t)l * 128 * 128, 128, 128);
        pack(pnW1 + (size_t)l * 128 * 128, pnW1p + (size_t)l * 128 * 128, 128, 128);
        pack(pnW2 + (size_t)l * 128 * 128, pnW2p + (size_t)l * 128 * 128, 128, 128);
    }
    pack(dW0, dW0p, 128, 128); pack(dW1, dW1p, 128, 128);

    // ---- state init ----
    (void)hipMemcpyAsync(posB, coords, (size_t)GNS_N * 3 * 4, hipMemcpyDeviceToDevice, stream);
    (void)hipMemcpyAsync(velB, vels,   (size_t)GNS_N * 3 * 4, hipMemcpyDeviceToDevice, stream);
    hipLaunchKernelGGL(gns_f32_to_f16, dim3((GNS_N * 32 + 255) / 256), dim3(256), 0, stream,
                       xin, x16, GNS_N * 32);

    // ---- node encoder (pos-independent; compute once) ----
    launch_gemm(x16,   nW0p, nb0, nullptr, nullptr, nullptr, nullptr, nh16a, GNS_N, 32, 0, stream);
    launch_gemm(nh16a, nW1p, nb1, nullptr, nullptr, nullptr, nullptr, nh16b, GNS_N, 128, 0, stream);
    launch_gemm(nh16b, nW2p, nb2, ng, nbeta, nullptr, xl0_32, xl0_16, GNS_N, 128, 1, stream);

    for (int step = 0; step < GNS_STEPS; ++step) {
        // KNN graph + edge features
        hipLaunchKernelGGL(gns_knn, dim3(GNS_N), dim3(256), 0, stream, posB, snd, rcv);
        hipLaunchKernelGGL(gns_edge_attr, dim3((GNS_E + 255) / 256), dim3(256), 0, stream,
                           posB, resn, snd, rcv, ea16);
        // edge encoder
        launch_gemm(ea16,  eW0p, eb0, nullptr, nullptr, nullptr, nullptr, eh16a, GNS_E, 32, 0, stream);
        launch_gemm(eh16a, eW1p, eb1, nullptr, nullptr, nullptr, nullptr, eh16b, GNS_E, 128, 0, stream);
        launch_gemm(eh16b, eW2p, eb2, eg, ebeta, nullptr, nullptr, e16, GNS_E, 128, 1, stream);

        // xl <- xl0 (node encoding, recomputed each step in the reference)
        (void)hipMemcpyAsync(xlA32, xl0_32, (size_t)GNS_N * 128 * 4, hipMemcpyDeviceToDevice, stream);
        (void)hipMemcpyAsync(xlA16, xl0_16, (size_t)GNS_N * 128 * 2, hipMemcpyDeviceToDevice, stream);
        float* curX32 = xlA32; _Float16* curX16 = xlA16;
        float* nxtX32 = xlB32; _Float16* nxtX16 = xlB16;

        for (int l = 0; l < GNS_L; ++l) {
            // m = MLP_LN([x[rcv], x[snd], e])
            hipLaunchKernelGGL(gns_concat, dim3(GNS_E), dim3(128), 0, stream,
                               curX16, e16, snd, rcv, xcat16);
            launch_gemm(xcat16, peW0p + (size_t)l * 384 * 128, peb0 + l * 128,
                        nullptr, nullptr, nullptr, nullptr, eh16a, GNS_E, 384, 0, stream);
            launch_gemm(eh16a, peW1p + (size_t)l * 128 * 128, peb1 + l * 128,
                        nullptr, nullptr, nullptr, nullptr, eh16b, GNS_E, 128, 0, stream);
            launch_gemm(eh16b, peW2p + (size_t)l * 128 * 128, peb2 + l * 128,
                        peg + l * 128, pebeta + l * 128, nullptr, m32, e16, GNS_E, 128, 1, stream);
            // agg = m.view(N,128,K).sum(-1)
            hipLaunchKernelGGL(gns_aggregate, dim3((GNS_N * 128 + 255) / 256), dim3(256), 0, stream,
                               m32, agg16);
            // x = MLP_LN(agg) + x
            launch_gemm(agg16, pnW0p + (size_t)l * 128 * 128, pnb0 + l * 128,
                        nullptr, nullptr, nullptr, nullptr, nh16a, GNS_N, 128, 0, stream);
            launch_gemm(nh16a, pnW1p + (size_t)l * 128 * 128, pnb1 + l * 128,
                        nullptr, nullptr, nullptr, nullptr, nh16b, GNS_N, 128, 0, stream);
            launch_gemm(nh16b, pnW2p + (size_t)l * 128 * 128, pnb2 + l * 128,
                        png + l * 128, pnbeta + l * 128, curX32, nxtX32, nxtX16, GNS_N, 128, 1, stream);
            { float* t32 = curX32; curX32 = nxtX32; nxtX32 = t32;
              _Float16* t16 = curX16; curX16 = nxtX16; nxtX16 = t16; }
        }

        // decoder + Euler update
        launch_gemm(curX16, dW0p, db0, nullptr, nullptr, nullptr, nullptr, nh16a, GNS_N, 128, 0, stream);
        launch_gemm(nh16a,  dW1p, db1, nullptr, nullptr, nullptr, nh32, nullptr, GNS_N, 128, 0, stream);
        hipLaunchKernelGGL(gns_decode_update, dim3((GNS_N + 255) / 256), dim3(256), 0, stream,
                           nh32, dW2, db2, posB, velB);
    }

    (void)hipMemcpyAsync(d_out, posB, (size_t)GNS_N * 3 * 4, hipMemcpyDeviceToDevice, stream);
}
